// CausalSelfAttention_52819507806742
// MI455X (gfx1250) — matrix operand order repaired and checked by reading, not compiled, not run on hardware
//
#include <hip/hip_runtime.h>

// ---------------- types / helpers ----------------
typedef __bf16          v16bf __attribute__((ext_vector_type(16)));
typedef float           v8f   __attribute__((ext_vector_type(8)));
typedef unsigned int    v4u   __attribute__((ext_vector_type(4)));
typedef unsigned short  u16;

#define LOG2E 1.44269504088896340736f
#define EXP2F(x) __builtin_amdgcn_exp2f(x)   // v_exp_f32, no libm

__device__ __forceinline__ u16 f2bf(float f) {
  unsigned u = __float_as_uint(f);
  u += 0x7fffu + ((u >> 16) & 1u);           // round-to-nearest-even
  return (u16)(u >> 16);
}

// Build a 16xbf16 fragment from two 16-byte chunks (global or LDS).
__device__ __forceinline__ v16bf load_frag(const u16* p0, const u16* p1) {
  union { v4u q[2]; v16bf v; } u;
  u.q[0] = *(const v4u*)p0;
  u.q[1] = *(const v4u*)p1;
  return u.v;
}

__device__ __forceinline__ v8f wmma_bf16(v16bf a, v16bf b, v8f c) {
  return __builtin_amdgcn_wmma_f32_16x16x32_bf16(false, a, false, b, (short)0, c, false, false);
}

// Async 16B global->LDS copy (CDNA5, tracked by ASYNCcnt).
// LDS operand = byte offset within workgroup LDS (low 32 bits of flat shared addr).
__device__ __forceinline__ void async_g2l_b128(const u16* gptr, u16* lptr) {
  unsigned loff = (unsigned)(unsigned long long)(void*)lptr;
  asm volatile("global_load_async_to_lds_b128 %0, %1, off"
               :: "v"(loff), "v"(gptr) : "memory");
}
#define WAIT_ASYNC_0 asm volatile("s_wait_asynccnt 0" ::: "memory")
#define WAIT_ASYNC_4 asm volatile("s_wait_asynccnt 4" ::: "memory")
#define WAIT_ASYNC_6 asm volatile("s_wait_asynccnt 6" ::: "memory")

// xor-shuffle inside 16-lane group via ds_swizzle (group-of-32 encoding: xor<<10 | and=0x1f)
#define DSWZ(v, xm) __int_as_float(__builtin_amdgcn_ds_swizzle(__float_as_int(v), (((xm) << 10) | 0x1f)))

__device__ __forceinline__ float redmax16(float v) {
  v = fmaxf(v, DSWZ(v, 1)); v = fmaxf(v, DSWZ(v, 2));
  v = fmaxf(v, DSWZ(v, 4)); v = fmaxf(v, DSWZ(v, 8));
  return v;
}
__device__ __forceinline__ float redsum16(float v) {
  v += DSWZ(v, 1); v += DSWZ(v, 2); v += DSWZ(v, 4); v += DSWZ(v, 8);
  return v;
}

// ---------------- fp32 -> bf16 convert ----------------
__global__ __launch_bounds__(256) void k_cvt_bf16(const float* __restrict__ in,
                                                  u16* __restrict__ out, int n) {
  int i = blockIdx.x * 256 + threadIdx.x;
  if (i < n) out[i] = f2bf(in[i]);
}

// ---------------- shared WMMA GEMM core with async-LDS double buffering ----------------
// Block tile 128M x 256N, 8 waves each 64x64 (4x4 wmma tiles), K-step 32.
// LDS: A 2x[128x32] bf16 (16KB), B 2x[256x32] bf16 (32KB). 6 async copies per tile.
__device__ __forceinline__ void gemm_core_async(const u16* __restrict__ A,
                                                const u16* __restrict__ W,
                                                int K, int Mb, int Nb, int tid,
                                                u16* shA, u16* shB, v8f acc[4][4]) {
  const int lane = tid & 31, wave = tid >> 5;
  const int g = lane >> 4, ln = lane & 15;
  const int wm = (wave & 1) * 64, wn = (wave >> 1) * 64;

  v8f z = {0.f, 0.f, 0.f, 0.f, 0.f, 0.f, 0.f, 0.f};
#pragma unroll
  for (int mi = 0; mi < 4; ++mi)
#pragma unroll
    for (int nj = 0; nj < 4; ++nj) acc[mi][nj] = z;

  auto issue = [&](int kk, int buf) {
    u16* ad = shA + buf * (128 * 32);
#pragma unroll
    for (int it = 0; it < 2; ++it) {            // 512 16B chunks / 256 threads
      int ch = tid + 256 * it, r = ch >> 2, c = ch & 3;
      async_g2l_b128(A + (size_t)(Mb + r) * K + kk + c * 8, ad + ch * 8);
    }
    u16* bd = shB + buf * (256 * 32);
#pragma unroll
    for (int it = 0; it < 4; ++it) {            // 1024 chunks / 256 threads
      int ch = tid + 256 * it, r = ch >> 2, c = ch & 3;
      async_g2l_b128(W + (size_t)(Nb + r) * K + kk + c * 8, bd + ch * 8);
    }
  };

  issue(0, 0);
  for (int kk = 0; kk < K; kk += 32) {
    int cur = (kk >> 5) & 1;
    if (kk + 32 < K) { issue(kk + 32, cur ^ 1); WAIT_ASYNC_6; }  // in-order: drains current tile
    else            { WAIT_ASYNC_0; }
    __syncthreads();                             // tile visible to all waves

    const u16* ac = shA + cur * (128 * 32);
    const u16* bc = shB + cur * (256 * 32);
    v16bf af[4];
#pragma unroll
    for (int mi = 0; mi < 4; ++mi) {             // A: K runs [8g,+8) and [16+8g,+8)
      const u16* ar = ac + (wm + 16 * mi + ln) * 32;
      af[mi] = load_frag(ar + 8 * g, ar + 16 + 8 * g);
    }
#pragma unroll
    for (int nj = 0; nj < 4; ++nj) {             // B: K runs [8g,+8) and [16+8g,+8)
      const u16* br = bc + (wn + 16 * nj + ln) * 32;
      v16bf bf = load_frag(br + 8 * g, br + 16 + 8 * g);
#pragma unroll
      for (int mi = 0; mi < 4; ++mi)
        acc[mi][nj] = wmma_bf16(af[mi], bf, acc[mi][nj]);
    }
    __syncthreads();                             // protect LDS before next tile's copies land
  }
}

// ---------------- GEMM 1: qkv = x @ Wqkv^T + b ; scatter to Q/K/V buffers ----------------
// Q,K: [B*H, T, D] bf16 ; V: transposed [B*H, D, T] bf16 (so PV B-fragments are contiguous)
__global__ __launch_bounds__(256) void k_gemm_qkv(const u16* __restrict__ A,
                                                  const u16* __restrict__ W,
                                                  const float* __restrict__ bias,
                                                  u16* __restrict__ q_o,
                                                  u16* __restrict__ k_o,
                                                  u16* __restrict__ vt_o) {
  __shared__ u16 shA[2 * 128 * 32];
  __shared__ u16 shB[2 * 256 * 32];
  const int tid = threadIdx.x;
  const int lane = tid & 31, wave = tid >> 5;
  const int g = lane >> 4, ln = lane & 15;
  const int Mb = blockIdx.x * 128, Nb = blockIdx.y * 256;

  v8f acc[4][4];
  gemm_core_async(A, W, 1024, Mb, Nb, tid, shA, shB, acc);

  const int wm = Mb + (wave & 1) * 64, wn = Nb + (wave >> 1) * 64;
#pragma unroll
  for (int mi = 0; mi < 4; ++mi)
#pragma unroll
    for (int nj = 0; nj < 4; ++nj) {
      int ncol = wn + nj * 16 + ln;              // [0,3072)
      float bv = bias[ncol];
      int sel = ncol >> 10;                      // 0=q 1=k 2=v
      int cc = ncol & 1023, hh = cc >> 6, dd = cc & 63;
#pragma unroll
      for (int i = 0; i < 8; ++i) {
        int grow = wm + mi * 16 + i + 8 * g;     // global token row (C layout)
        int b_ = grow >> 11, t_ = grow & 2047;
        u16 val = f2bf(acc[mi][nj][i] + bv);
        size_t bh = (size_t)(b_ * 16 + hh);
        if (sel == 0)      q_o[(bh * 2048 + t_) * 64 + dd] = val;
        else if (sel == 1) k_o[(bh * 2048 + t_) * 64 + dd] = val;
        else               vt_o[(bh * 64 + dd) * 2048 + t_] = val;
      }
    }
}

// ---------------- Flash-style causal attention ----------------
// grid: (qblocks=32, B*H=64), 128 threads (4 waves). Each wave: 16 q-rows, full D=64.
// K/V tiles are shared by all 4 waves -> staged once per block via async LDS copies.
__global__ __launch_bounds__(128) void k_attn(const u16* __restrict__ qb,
                                              const u16* __restrict__ kbf,
                                              const u16* __restrict__ vtb,
                                              u16* __restrict__ attb) {
  __shared__ u16 shK[2 * 32 * 64];               // [kj 32][d 64] per buffer
  __shared__ u16 shV[2 * 64 * 32];               // [d 64][kj 32] per buffer
  __shared__ u16 ldsP[4][16 * 32];               // per-wave P staging (C-layout -> A-frag)

  const int tid = threadIdx.x;
  const int lane = tid & 31, wave = tid >> 5;
  const int g = lane >> 4, ln = lane & 15;
  const int qblk = blockIdx.x, bh = blockIdx.y;
  const int q0 = qblk * 64 + wave * 16;
  const float sc = 0.125f * LOG2E;               // 1/sqrt(64) folded with log2(e)

  // Q A-fragments (d 0..31 and 32..63), loaded once
  const u16* qrow = qb + ((size_t)bh * 2048 + q0 + ln) * 64;
  v16bf qa0 = load_frag(qrow + 8 * g, qrow + 16 + 8 * g);
  v16bf qa1 = load_frag(qrow + 32 + 8 * g, qrow + 48 + 8 * g);

  v8f z = {0.f, 0.f, 0.f, 0.f, 0.f, 0.f, 0.f, 0.f};
  v8f o[4] = {z, z, z, z};
  float m8[8], l8[8];
#pragma unroll
  for (int i = 0; i < 8; ++i) { m8[i] = -1e30f; l8[i] = 0.f; }

  u16* lp = &ldsP[wave][0];

  auto issue = [&](int t, int buf) {
    int kb0 = t * 32;
    u16* kd = shK + buf * (32 * 64);
#pragma unroll
    for (int it = 0; it < 2; ++it) {             // K tile: 256 chunks / 128 threads
      int ch = tid + 128 * it, r = ch >> 3, c = ch & 7;
      async_g2l_b128(kbf + ((size_t)bh * 2048 + kb0 + r) * 64 + c * 8, kd + ch * 8);
    }
    u16* vd = shV + buf * (64 * 32);
#pragma unroll
    for (int it = 0; it < 2; ++it) {             // V tile: 256 chunks / 128 threads
      int ch = tid + 128 * it, r = ch >> 2, c = ch & 3;
      async_g2l_b128(vtb + ((size_t)bh * 64 + r) * 2048 + kb0 + c * 8, vd + ch * 8);
    }
  };

  const int ntiles = (qblk * 64 + 64) >> 5;      // causal: only tiles up to the diagonal
  issue(0, 0);
  for (int t = 0; t < ntiles; ++t) {
    int cur = t & 1;
    if (t + 1 < ntiles) { issue(t + 1, cur ^ 1); WAIT_ASYNC_4; }
    else                { WAIT_ASYNC_0; }
    __syncthreads();

    const u16* kc = shK + cur * (32 * 64);
    const u16* vc = shV + cur * (64 * 32);
    const int kb0 = t * 32;

    // ---- S = Q K^T (two 16x16 tiles over kj, K-dim = d split 2x32) ----
    const u16* k0r = kc + ln * 64;
    const u16* k1r = kc + (16 + ln) * 64;
    v16bf kb00 = load_frag(k0r + 8 * g, k0r + 16 + 8 * g);
    v16bf kb01 = load_frag(k0r + 32 + 8 * g, k0r + 48 + 8 * g);
    v16bf kb10 = load_frag(k1r + 8 * g, k1r + 16 + 8 * g);
    v16bf kb11 = load_frag(k1r + 32 + 8 * g, k1r + 48 + 8 * g);

    v8f s0 = wmma_bf16(qa0, kb00, z); s0 = wmma_bf16(qa1, kb01, s0);
    v8f s1 = wmma_bf16(qa0, kb10, z); s1 = wmma_bf16(qa1, kb11, s1);

    const int c0 = kb0 + ln, c1 = kb0 + 16 + ln;

    // ---- online softmax (per row i+8g; stats uniform within 16-lane group) ----
#pragma unroll
    for (int i = 0; i < 8; ++i) {
      int row = q0 + i + 8 * g;
      float v0 = (c0 <= row) ? s0[i] * sc : -3.0e38f;
      float v1 = (c1 <= row) ? s1[i] * sc : -3.0e38f;
      float rmax = redmax16(fmaxf(v0, v1));
      float mnew = fmaxf(m8[i], rmax);
      float alpha = EXP2F(m8[i] - mnew);
      m8[i] = mnew;
      float p0 = EXP2F(v0 - mnew);
      float p1 = EXP2F(v1 - mnew);
      l8[i] = l8[i] * alpha + redsum16(p0 + p1);
#pragma unroll
      for (int j = 0; j < 4; ++j) o[j][i] *= alpha;
      // stash P (bf16) in LDS, row-major 16x32 per wave
      lp[(i + 8 * g) * 32 + ln]      = f2bf(p0);
      lp[(i + 8 * g) * 32 + 16 + ln] = f2bf(p1);
    }
    asm volatile("s_wait_dscnt 0x0" ::: "memory");  // same-wave LDS RAW (in-order DS)

    // ---- P as A-fragment (16 x 32 over kj) ----
    v16bf pa = load_frag(lp + ln * 32 + 8 * g, lp + ln * 32 + 16 + 8 * g);

    // ---- O += P V (one wmma per 16-wide d tile; K-dim = 32 kj) ----
#pragma unroll
    for (int j = 0; j < 4; ++j) {
      const u16* vr = vc + (j * 16 + ln) * 32;
      v16bf vb = load_frag(vr + 8 * g, vr + 16 + 8 * g);
      o[j] = wmma_bf16(pa, vb, o[j]);
    }
    __syncthreads();                             // all reads done before next tile lands
  }

  // ---- normalize and write bf16 attention output [B,T,C] ----
  const int bi = bh >> 4, hh = bh & 15;
#pragma unroll
  for (int i = 0; i < 8; ++i) {
    float inv = 1.0f / l8[i];
    int q = q0 + i + 8 * g;
    size_t rowbase = ((size_t)bi * 2048 + q) * 1024 + hh * 64;
#pragma unroll
    for (int j = 0; j < 4; ++j)
      attb[rowbase + j * 16 + ln] = f2bf(o[j][i] * inv);
  }
}

// ---------------- GEMM 2: out = att @ Wproj^T + b (fp32 out) ----------------
__global__ __launch_bounds__(256) void k_gemm_proj(const u16* __restrict__ A,
                                                   const u16* __restrict__ W,
                                                   const float* __restrict__ bias,
                                                   float* __restrict__ out) {
  __shared__ u16 shA[2 * 128 * 32];
  __shared__ u16 shB[2 * 256 * 32];
  const int tid = threadIdx.x;
  const int lane = tid & 31, wave = tid >> 5;
  const int g = lane >> 4, ln = lane & 15;
  const int Mb = blockIdx.x * 128, Nb = blockIdx.y * 256;

  v8f acc[4][4];
  gemm_core_async(A, W, 1024, Mb, Nb, tid, shA, shB, acc);

  const int wm = Mb + (wave & 1) * 64, wn = Nb + (wave >> 1) * 64;
#pragma unroll
  for (int mi = 0; mi < 4; ++mi)
#pragma unroll
    for (int nj = 0; nj < 4; ++nj) {
      int ncol = wn + nj * 16 + ln;
      float bv = bias[ncol];
#pragma unroll
      for (int i = 0; i < 8; ++i) {
        int grow = wm + mi * 16 + i + 8 * g;
        out[(size_t)grow * 1024 + ncol] = acc[mi][nj][i] + bv;
      }
    }
}

// ---------------- launcher ----------------
extern "C" void kernel_launch(void* const* d_in, const int* in_sizes, int n_in,
                              void* d_out, int out_size, void* d_ws, size_t ws_size,
                              hipStream_t stream) {
  (void)in_sizes; (void)n_in; (void)out_size; (void)ws_size;
  const float* x      = (const float*)d_in[0];
  const float* w_qkv  = (const float*)d_in[1];
  const float* b_qkv  = (const float*)d_in[2];
  const float* w_proj = (const float*)d_in[3];
  const float* b_proj = (const float*)d_in[4];
  float* out = (float*)d_out;

  u16* base = (u16*)d_ws;
  size_t off = 0;
  auto carve = [&](size_t n) { u16* p = base + off; off += (n + 127) & ~(size_t)127; return p; };
  u16* xb     = carve((size_t)8192 * 1024);     // x bf16
  u16* wqkvb  = carve((size_t)3072 * 1024);     // Wqkv bf16
  u16* wprojb = carve((size_t)1024 * 1024);     // Wproj bf16
  u16* qb     = carve((size_t)64 * 2048 * 64);  // [BH,T,D]
  u16* kbuf   = carve((size_t)64 * 2048 * 64);  // [BH,T,D]
  u16* vtb    = carve((size_t)64 * 64 * 2048);  // [BH,D,T]
  u16* attb   = carve((size_t)8192 * 1024);     // [B,T,C] bf16

  k_cvt_bf16<<<8388608 / 256, 256, 0, stream>>>(x, xb, 8388608);
  k_cvt_bf16<<<3145728 / 256, 256, 0, stream>>>(w_qkv, wqkvb, 3145728);
  k_cvt_bf16<<<1048576 / 256, 256, 0, stream>>>(w_proj, wprojb, 1048576);

  k_gemm_qkv<<<dim3(64, 12), 256, 0, stream>>>(xb, wqkvb, b_qkv, qb, kbuf, vtb);
  k_attn<<<dim3(32, 64), 128, 0, stream>>>(qb, kbuf, vtb, attb);
  k_gemm_proj<<<dim3(64, 4), 256, 0, stream>>>(attb, wprojb, b_proj, out);
}
